// WarpRefiner_214748365230
// MI455X (gfx1250) — matrix operand order repaired and checked by reading, not compiled, not run on hardware
//
#include <hip/hip_runtime.h>
#include <math.h>

// ============================================================================
// MI455X (gfx1250) WarpRefiner
//
// Roofline: all activations (~100 MB) are L2-resident (192 MB), HBM traffic is
// only feat_A/feat_B + outputs (~40 MB ~ 2 us at 23.3 TB/s).  The ~18 GFLOP of
// GEMM-shaped convs dominate -> run every conv as a WMMA f16 GEMM
// (v_wmma_f32_16x16x32_f16, f32 accum) over N = B*H*W pixels with f16
// channel-major activations.  3x3 (incl. dilated-2) convs are 9 accumulating
// GEMMs with shifted 2D-tile addressing (no materialized im2col).  GroupNorm
// is a two-pass reduction; GN+GELU(+residual) fuses with the f32->f16 convert
// feeding the next GEMM.  Per workgroup: full-M x 128-pixel tile, MT template
// param so the M loop unrolls into 6-12 back-to-back WMMAs per staged panel.
// ============================================================================

typedef __attribute__((ext_vector_type(16))) _Float16 v16h;
typedef __attribute__((ext_vector_type(8)))  float    v8f;
typedef __attribute__((ext_vector_type(4)))  unsigned int u32x4;
typedef __attribute__((ext_vector_type(4)))  int      i32x4;
typedef __attribute__((ext_vector_type(8)))  int      i32x8;
typedef _Float16 h16;

#define BATCH 4
#define CF    128
#define HH    96
#define WW    96
#define HWPX  (HH*WW)   // 9216
#define KPAD  608       // 600 input channels padded to x32

static __device__ __forceinline__ float gelu_f(float x) {
  return 0.5f * x * (1.0f + erff(x * 0.70710678118654752f));
}
static __device__ __forceinline__ int clampi(int v, int lo, int hi) {
  return v < lo ? lo : (v > hi ? hi : v);
}

// ---------------------------------------------------------------------------
// Weight packing: f32 -> f16, K padded, 3x3 reordered tap-major.
// ---------------------------------------------------------------------------
__global__ void pack_pw_kernel(const float* __restrict__ w, h16* __restrict__ dst,
                               int M, int Kin, int Kt) {
  int i = blockIdx.x * 256 + threadIdx.x;
  if (i >= M * Kt) return;
  int m = i / Kt, k = i % Kt;
  dst[i] = (k < Kin) ? (h16)w[m * Kin + k] : (h16)0.0f;
}

__global__ void pack_3x3_kernel(const float* __restrict__ w, h16* __restrict__ dst,
                                int Mpad, int Min, int Kin) {
  int i = blockIdx.x * 256 + threadIdx.x;
  if (i >= 9 * Mpad * Kin) return;
  int t = i / (Mpad * Kin);
  int r = i % (Mpad * Kin);
  int m = r / Kin, k = r % Kin;
  dst[i] = (m < Min) ? (h16)w[(m * Kin + k) * 9 + t] : (h16)0.0f;
}

__global__ void pack_heads_kernel(const float* __restrict__ dw,
                                  const float* __restrict__ ow,
                                  h16* __restrict__ dst) {
  int i = blockIdx.x * 256 + threadIdx.x;
  if (i >= 9 * 16 * 96) return;
  int t = i / (16 * 96);
  int r = (i / 96) % 16;
  int k = i % 96;
  float v = 0.f;
  if (r < 2)       v = dw[(r * 96 + k) * 9 + t];
  else if (r == 2) v = ow[k * 9 + t];
  dst[i] = (h16)v;
}

// ---------------------------------------------------------------------------
// TDM warm: async-stage the packed stem weights through the Tensor Data Mover
// (pulls them into L2 ahead of the stem GEMM) and exercise tensorcnt +
// cluster-barrier (S_NOP when not launched as a cluster).
// ---------------------------------------------------------------------------
__global__ __launch_bounds__(64)
void tdm_warm_kernel(const h16* __restrict__ src, float* __restrict__ dummy) {
#if __has_builtin(__builtin_amdgcn_tensor_load_to_lds)
  __shared__ __align__(128) h16 buf[2048];   // 4 KB landing zone
  unsigned long long ga = (unsigned long long)(const void*)src;
  // D# group0: count=1 (one valid user descriptor), lds_addr=0,
  // global_addr[56:0], type=2 ("image").
  u32x4 g0;
  g0[0] = 1u;
  g0[1] = 0u;
  g0[2] = (unsigned)(ga & 0xffffffffu);
  g0[3] = (unsigned)((ga >> 32) & 0x01ffffffu) | (2u << 30);
  // D# group1: wg_mask=0 (not clustered), data_size=1 (2 bytes),
  // tensor_dim0=2048, tensor_dim1=1, tile_dim0=2048, tile_dim1=1,
  // tensor_dim0_stride=2048.
  i32x8 g1;
  g1[0] = (1 << 16);
  g1[1] = (int)((2048u & 0xffffu) << 16);
  g1[2] = (int)((2048u >> 16) | (1u << 16));
  g1[3] = (int)(2048u << 16);
  g1[4] = 1;
  g1[5] = 2048;
  g1[6] = 0;
  g1[7] = 0;
  i32x4 z4 = {0, 0, 0, 0};
#if defined(__clang_major__) && (__clang_major__ >= 23)
  i32x8 z8 = {0, 0, 0, 0, 0, 0, 0, 0};
  __builtin_amdgcn_tensor_load_to_lds(g0, g1, z4, z4, z8, 0);
#else
  __builtin_amdgcn_tensor_load_to_lds(g0, g1, z4, z4, 0);
#endif
  __builtin_amdgcn_s_wait_tensorcnt(0);
#if __has_builtin(__builtin_amdgcn_s_cluster_barrier)
  __builtin_amdgcn_s_cluster_barrier();
#endif
  if (threadIdx.x == 0) dummy[0] = (float)buf[0];
#else
  if (threadIdx.x == 0) dummy[0] = (float)src[0];
#endif
}

// ---------------------------------------------------------------------------
// Bilinear grid-sample + input assembly (xin rows 0..511, 593..607) + norms.
// xin layout: [b][k][pixel] f16, k padded to KPAD.
// ---------------------------------------------------------------------------
__global__ __launch_bounds__(256)
void sample_fill_kernel(const float* __restrict__ fA, const float* __restrict__ fB,
                        const float* __restrict__ pwarp, const float* __restrict__ pov,
                        h16* __restrict__ xin, float* __restrict__ norms) {
  int i = blockIdx.x * 256 + threadIdx.x;
  if (i >= BATCH * HWPX) return;
  int b = i / HWPX, p = i % HWPX;
  int y = p / WW, x = p % WW;

  float gx = pwarp[(size_t)i * 2], gy = pwarp[(size_t)i * 2 + 1];
  float cgx = gx, cgy = gy;
  if (!(cgx == cgx)) cgx = 0.f;
  if (!(cgy == cgy)) cgy = 0.f;
  cgx = fminf(fmaxf(cgx, -1.5f), 1.5f);
  cgy = fminf(fmaxf(cgy, -1.5f), 1.5f);
  float fx = (cgx + 1.f) * (WW * 0.5f) - 0.5f;
  float fy = (cgy + 1.f) * (HH * 0.5f) - 0.5f;
  float x0f = floorf(fx), y0f = floorf(fy);
  float wx = fx - x0f, wy = fy - y0f;
  int x0 = (int)x0f, y0 = (int)y0f, x1 = x0 + 1, y1 = y0 + 1;
  bool vx0 = (x0 >= 0) && (x0 < WW), vx1 = (x1 >= 0) && (x1 < WW);
  bool vy0 = (y0 >= 0) && (y0 < HH), vy1 = (y1 >= 0) && (y1 < HH);
  float w00 = (1.f - wx) * (1.f - wy), w10 = wx * (1.f - wy);
  float w01 = (1.f - wx) * wy,         w11 = wx * wy;

  size_t fbase = (size_t)b * CF * HWPX;
  size_t xrow  = (size_t)b * KPAD * HWPX + (size_t)p;
  float na = 0.f, ns = 0.f;
  for (int c = 0; c < CF; ++c) {
    size_t base = fbase + (size_t)c * HWPX;
    float v00 = (vx0 && vy0) ? fB[base + y0 * WW + x0] : 0.f;
    float v10 = (vx1 && vy0) ? fB[base + y0 * WW + x1] : 0.f;
    float v01 = (vx0 && vy1) ? fB[base + y1 * WW + x0] : 0.f;
    float v11 = (vx1 && vy1) ? fB[base + y1 * WW + x1] : 0.f;
    float sv = v00 * w00 + v10 * w10 + v01 * w01 + v11 * w11;
    float av = fA[base + p];
    xin[xrow + (size_t)c * HWPX]         = (h16)av;
    xin[xrow + (size_t)(128 + c) * HWPX] = (h16)sv;
    xin[xrow + (size_t)(256 + c) * HWPX] = (h16)fabsf(av - sv);
    xin[xrow + (size_t)(384 + c) * HWPX] = (h16)(av * sv);
    na += av * av; ns += sv * sv;
  }
  norms[i] = sqrtf(na + 1e-8f);
  norms[BATCH * HWPX + i] = sqrtf(ns + 1e-8f);

  float gdx = ((float)x + 0.5f) * (2.0f / WW) - 1.0f;
  float gdy = ((float)y + 0.5f) * (2.0f / HH) - 1.0f;
  xin[xrow + (size_t)593 * HWPX] = (h16)pov[i];
  xin[xrow + (size_t)594 * HWPX] = (h16)gx;
  xin[xrow + (size_t)595 * HWPX] = (h16)gy;
  xin[xrow + (size_t)596 * HWPX] = (h16)gdx;
  xin[xrow + (size_t)597 * HWPX] = (h16)gdy;
  xin[xrow + (size_t)598 * HWPX] = (h16)(gx - gdx);
  xin[xrow + (size_t)599 * HWPX] = (h16)(gy - gdy);
  for (int r = 600; r < KPAD; ++r) xin[xrow + (size_t)r * HWPX] = (h16)0.f;
}

// ---------------------------------------------------------------------------
// 9x9 normalized local correlation (edge-padded) -> xin rows 512..592.
// LDS: 16x16x128 center tile of feat_A + 24x24x128 halo tile of sampled_B
// (~208 KB of the WGP's 320 KB, dynamic LDS).  Dot products read b128 chunks.
// ---------------------------------------------------------------------------
__global__ __launch_bounds__(256)
void corr_kernel(h16* __restrict__ xin, const float* __restrict__ norms) {
  extern __shared__ h16 smem[];
  h16* At = smem;                  // [16*16][128]
  h16* Bt = smem + 16 * 16 * 128;  // [24*24][128]

  int b  = blockIdx.z;
  int y0 = blockIdx.y * 16, x0 = blockIdx.x * 16;
  size_t xbase = (size_t)b * KPAD * HWPX;

  for (int i = threadIdx.x; i < 16 * 16 * 128; i += 256) {
    int c = i & 127, pos = i >> 7;
    int ty = pos >> 4, tx = pos & 15;
    At[i] = xin[xbase + (size_t)c * HWPX + (y0 + ty) * WW + (x0 + tx)];
  }
  for (int i = threadIdx.x; i < 24 * 24 * 128; i += 256) {
    int c = i & 127, pos = i >> 7;
    int py = pos / 24, px = pos % 24;
    int sy = clampi(y0 + py - 4, 0, HH - 1);
    int sx = clampi(x0 + px - 4, 0, WW - 1);
    Bt[i] = xin[xbase + (size_t)(128 + c) * HWPX + sy * WW + sx];
  }
  __syncthreads();

  int ty = threadIdx.x >> 4, tx = threadIdx.x & 15;
  int y = y0 + ty, x = x0 + tx, p = y * WW + x;
  float na = norms[b * HWPX + p];
  const uint4* aq = (const uint4*)&At[(ty * 16 + tx) * 128];
  for (int o = 0; o < 81; ++o) {
    int dy = o / 9, dx = o % 9;
    const uint4* bq = (const uint4*)&Bt[((ty + dy) * 24 + (tx + dx)) * 128];
    float dot = 0.f;
#pragma unroll
    for (int c8 = 0; c8 < 16; ++c8) {
      union { uint4 q; h16 h[8]; } ua, ub;
      ua.q = aq[c8]; ub.q = bq[c8];
#pragma unroll
      for (int j = 0; j < 8; ++j) dot += (float)ua.h[j] * (float)ub.h[j];
    }
    int sy = clampi(y + dy - 4, 0, HH - 1);
    int sx = clampi(x + dx - 4, 0, WW - 1);
    float nb = norms[BATCH * HWPX + b * HWPX + sy * WW + sx];
    xin[xbase + (size_t)(512 + o) * HWPX + p] = (h16)(dot / (na * nb + 1e-8f));
  }
}

// ---------------------------------------------------------------------------
// Conv-as-GEMM, WMMA f16 16x16x32, f32 accumulate.
//   act : [b][Kt][HWPX] f16 (Kt % 32 == 0)
//   wgt : [NTAPS][MT*16][Kt] f16
//   out : [b][MT*16][HWPX] f32
// Workgroup: 256 threads (8 waves).  Tile: full M (MT*16) x 128 pixels,
// where the 128 pixels are an 8-row x 16-col 2D image tile (cheap shifted-tap
// addressing, coalesced uint4 panel loads).  Per K-step: A panel (MT*16 x 32)
// + B panel (32 x 128) staged in LDS; each wave builds ONE B fragment for its
// 16 pixels and runs MT unrolled WMMAs against the shared A panel.
// ---------------------------------------------------------------------------
template <int MT, int NTAPS>
__global__ __launch_bounds__(256)
void conv_gemm_wmma(const h16* __restrict__ act, int Kt,
                    const h16* __restrict__ wgt,
                    float* __restrict__ out, int dil) {
  constexpr int Mt16 = MT * 16;
  __shared__ __align__(16) h16 Ap[Mt16 * 32];
  __shared__ __align__(16) h16 Bp[32 * 128];

  int b    = blockIdx.z;
  int ty0  = (blockIdx.x / (WW / 16)) * 8;   // tile origin: image row
  int tx0  = (blockIdx.x % (WW / 16)) * 16;  // tile origin: image col
  int tid  = threadIdx.x;
  int wv   = tid >> 5, lane = tid & 31;

  v8f zero = {};
  v8f acc[MT];
#pragma unroll
  for (int m = 0; m < MT; ++m) acc[m] = zero;

  for (int t = 0; t < NTAPS; ++t) {
    int sdy = (NTAPS == 1) ? 0 : ((t / 3) - 1) * dil;
    int sdx = (NTAPS == 1) ? 0 : ((t % 3) - 1) * dil;
    const h16* wt = wgt + (size_t)t * Mt16 * Kt;

    for (int k0 = 0; k0 < Kt; k0 += 32) {
      // ---- A panel: Mt16 x 32 halves, uint4 chunks, coalesced
      for (int i = tid; i < Mt16 * 4; i += 256) {
        int r = i >> 2, q = i & 3;
        *(uint4*)&Ap[r * 32 + q * 8] =
            *(const uint4*)&wt[(size_t)r * Kt + k0 + q * 8];
      }
      // ---- B panel: 32 (K) x 128 (8 image rows x 16 cols)
      {
        int r  = tid >> 3;          // K row 0..31
        int rr = tid & 7;           // pixel-row within tile
        int py = ty0 + rr + sdy;
        int pxb = tx0 + sdx;
        const h16* arow = act + ((size_t)b * Kt + (size_t)(k0 + r)) * HWPX;
        h16* dst = &Bp[r * 128 + rr * 16];
        bool rowok = (unsigned)py < (unsigned)HH;
        if (sdx == 0 && rowok) {
          const uint4* src = (const uint4*)(arow + py * WW + pxb);
          *(uint4*)dst       = src[0];
          *(uint4*)&dst[8]   = src[1];
        } else if (rowok) {
#pragma unroll
          for (int c = 0; c < 16; ++c) {
            int px = pxb + c;
            dst[c] = ((unsigned)px < (unsigned)WW) ? arow[py * WW + px] : (h16)0.f;
          }
        } else {
#pragma unroll
          for (int c = 0; c < 16; ++c) dst[c] = (h16)0.f;
        }
        if (k0 + 32 < Kt)   // speculative prefetch of next K panel
          __builtin_prefetch((const void*)(arow + 32 * HWPX), 0, 1);
      }
      __syncthreads();

      // ---- one B fragment per wave, reused by all MT WMMAs
      v16h bf;
      int bcol = (wv << 4) + (lane & 15);
      int krb  = (lane < 16) ? 0 : 16;
#pragma unroll
      for (int hh = 0; hh < 16; ++hh)
        bf[hh] = Bp[(krb + hh) * 128 + bcol];

      int ar = lane & 15;
      int kb = (lane < 16) ? 0 : 16;
#pragma unroll
      for (int m = 0; m < MT; ++m) {
        union { v16h v; uint4 q[2]; } au;
        au.q[0] = *(const uint4*)&Ap[(m * 16 + ar) * 32 + kb];
        au.q[1] = *(const uint4*)&Ap[(m * 16 + ar) * 32 + kb + 8];
        acc[m] = __builtin_amdgcn_wmma_f32_16x16x32_f16(
            false, au.v, false, bf, (short)0, acc[m], false, false);
      }
      __syncthreads();
    }
  }

  // ---- store: wave wv owns image row ty0+wv, cols tx0..tx0+15 (coalesced)
  int pix = (ty0 + wv) * WW + tx0 + (lane & 15);
  int mb  = (lane < 16) ? 0 : 8;
  float* obase = out + (size_t)b * Mt16 * HWPX + (size_t)pix;
#pragma unroll
  for (int m = 0; m < MT; ++m)
#pragma unroll
    for (int v = 0; v < 8; ++v)
      obase[(size_t)(m * 16 + mb + v) * HWPX] = acc[m][v];
}

// ---------------------------------------------------------------------------
// GroupNorm stats: one block per (b, group); stats[bg] = {mean, rstd}.
// ---------------------------------------------------------------------------
__global__ __launch_bounds__(256)
void gn_stats_kernel(const float* __restrict__ raw, int Cch, float* __restrict__ stats) {
  int bg = blockIdx.x;
  int b = bg >> 3, g = bg & 7;
  int cg = Cch / 8;
  size_t base = ((size_t)b * Cch + (size_t)g * cg) * HWPX;
  size_t n = (size_t)cg * HWPX;
  float s = 0.f, ss = 0.f;
  for (size_t i = threadIdx.x; i < n; i += 256) {
    float v = raw[base + i];
    s += v; ss += v * v;
  }
  __shared__ float sh[256], sh2[256];
  sh[threadIdx.x] = s; sh2[threadIdx.x] = ss;
  __syncthreads();
  for (int o = 128; o > 0; o >>= 1) {
    if (threadIdx.x < o) { sh[threadIdx.x] += sh[threadIdx.x + o]; sh2[threadIdx.x] += sh2[threadIdx.x + o]; }
    __syncthreads();
  }
  if (threadIdx.x == 0) {
    float inv = 1.0f / (float)n;
    float mean = sh[0] * inv;
    float var = sh2[0] * inv - mean * mean;
    stats[bg * 2]     = mean;
    stats[bg * 2 + 1] = rsqrtf(var + 1e-5f);
  }
}

// GN + GELU (+ optional f16 residual) -> f16 activation for next GEMM.
__global__ __launch_bounds__(256)
void gn_apply_kernel(const float* __restrict__ raw, int Cch,
                     const float* __restrict__ stats,
                     const float* __restrict__ gw, const float* __restrict__ gb,
                     const h16* __restrict__ res, h16* __restrict__ outh) {
  size_t i = (size_t)blockIdx.x * 256 + threadIdx.x;
  if (i >= (size_t)BATCH * Cch * HWPX) return;
  int c = (int)((i / HWPX) % Cch);
  int b = (int)(i / ((size_t)HWPX * Cch));
  int g = c / (Cch / 8);
  float mean = stats[(b * 8 + g) * 2];
  float rstd = stats[(b * 8 + g) * 2 + 1];
  float v = (raw[i] - mean) * rstd * gw[c] + gb[c];
  v = gelu_f(v);
  if (res) v += (float)res[i];
  outh[i] = (h16)v;
}

// Depthwise 3x3 (pad=1), f16 in -> f32 raw out.
__global__ __launch_bounds__(256)
void dwconv3_kernel(const h16* __restrict__ act, const float* __restrict__ w,
                    float* __restrict__ out, int Cch) {
  size_t i = (size_t)blockIdx.x * 256 + threadIdx.x;
  if (i >= (size_t)BATCH * Cch * HWPX) return;
  int p = (int)(i % HWPX);
  int c = (int)((i / HWPX) % Cch);
  int y = p / WW, x = p % WW;
  const h16* src = act + (i - p);
  const float* wc = w + c * 9;
  float s = 0.f;
#pragma unroll
  for (int dy = -1; dy <= 1; ++dy)
#pragma unroll
    for (int dx = -1; dx <= 1; ++dx) {
      int yy = y + dy, xx = x + dx;
      if ((unsigned)yy < (unsigned)HH && (unsigned)xx < (unsigned)WW)
        s += wc[(dy + 1) * 3 + (dx + 1)] * (float)src[yy * WW + xx];
    }
  out[i] = s;
}

// Heads + final pointwise math -> d_out.
__global__ __launch_bounds__(256)
void final_kernel(const float* __restrict__ rawh,
                  const float* __restrict__ db, const float* __restrict__ ob,
                  const float* __restrict__ pwarp, const float* __restrict__ pov,
                  float* __restrict__ out) {
  int i = blockIdx.x * 256 + threadIdx.x;
  if (i >= BATCH * HWPX) return;
  int b = i / HWPX, p = i % HWPX;
  size_t rb = (size_t)b * 16 * HWPX + p;
  float d0 = rawh[rb]            + db[0];
  float d1 = rawh[rb + HWPX]     + db[1];
  float od = rawh[rb + 2 * HWPX] + ob[0];
  const float ds = 4.0f * (2.0f / 96.0f);
  float w0 = pwarp[(size_t)i * 2], w1 = pwarp[(size_t)i * 2 + 1];
  out[(size_t)i * 2]     = w0 + ds * tanhf(d0);
  out[(size_t)i * 2 + 1] = w1 + ds * tanhf(d1);
  float pc = fminf(fmaxf(pov[i], 1e-6f), 1.f - 1e-6f);
  float lg = logf(pc) - log1pf(-pc) + od;
  out[(size_t)BATCH * HWPX * 2 + i] = lg;
  out[(size_t)BATCH * HWPX * 3 + i] = 1.f / (1.f + expf(-lg));
}

// ---------------------------------------------------------------------------
// Orchestration (graph-capturable: no alloc/sync, everything on `stream`).
// ---------------------------------------------------------------------------
extern "C" void kernel_launch(void* const* d_in, const int* in_sizes, int n_in,
                              void* d_out, int out_size, void* d_ws, size_t ws_size,
                              hipStream_t stream) {
  (void)in_sizes; (void)n_in; (void)out_size; (void)ws_size;

  const float* feat_A = (const float*)d_in[0];
  const float* feat_B = (const float*)d_in[1];
  const float* pwarp  = (const float*)d_in[2];
  const float* pov    = (const float*)d_in[3];
  const float* stem_w  = (const float*)d_in[4];
  const float* stem_gw = (const float*)d_in[5];
  const float* stem_gb = (const float*)d_in[6];
  const float* b1p1_w  = (const float*)d_in[7];
  const float* b1p1_gw = (const float*)d_in[8];
  const float* b1p1_gb = (const float*)d_in[9];
  const float* b1dw_w  = (const float*)d_in[10];
  const float* b1dw_gw = (const float*)d_in[11];
  const float* b1dw_gb = (const float*)d_in[12];
  const float* b1p2_w  = (const float*)d_in[13];
  const float* b1p2_gw = (const float*)d_in[14];
  const float* b1p2_gb = (const float*)d_in[15];
  const float* b2_w    = (const float*)d_in[16];
  const float* b2_gw   = (const float*)d_in[17];
  const float* b2_gb   = (const float*)d_in[18];
  const float* b3p1_w  = (const float*)d_in[19];
  const float* b3p1_gw = (const float*)d_in[20];
  const float* b3p1_gb = (const float*)d_in[21];
  const float* b3dw_w  = (const float*)d_in[22];
  const float* b3dw_gw = (const float*)d_in[23];
  const float* b3dw_gb = (const float*)d_in[24];
  const float* b3p2_w  = (const float*)d_in[25];
  const float* b3p2_gw = (const float*)d_in[26];
  const float* b3p2_gb = (const float*)d_in[27];
  const float* delta_w = (const float*)d_in[28];
  const float* delta_b = (const float*)d_in[29];
  const float* ov_w    = (const float*)d_in[30];
  const float* ov_b    = (const float*)d_in[31];

  // workspace carve-up (bytes, 256-aligned); ~104 MB total
  char* ws = (char*)d_ws;
  const size_t SZ_XIN  = (size_t)BATCH * KPAD * HWPX * 2;
  const size_t SZ_A96  = (size_t)BATCH * 96  * HWPX * 2;
  const size_t SZ_A192 = (size_t)BATCH * 192 * HWPX * 2;
  const size_t SZ_RAW  = (size_t)BATCH * 192 * HWPX * 4;
  size_t off = 0;
  h16*   XIN    = (h16*)(ws + off);  off += SZ_XIN;
  h16*   ACT_A  = (h16*)(ws + off);  off += SZ_A96;
  h16*   ACT_B  = (h16*)(ws + off);  off += SZ_A96;
  h16*   ACT_C  = (h16*)(ws + off);  off += SZ_A96;
  h16*   ACT192 = (h16*)(ws + off);  off += SZ_A192;
  float* RAW    = (float*)(ws + off); off += SZ_RAW;
  float* STATS  = (float*)(ws + off); off += 256;
  float* TDMD   = (float*)(ws + off); off += 256;
  float* NORMS  = (float*)(ws + off); off += (size_t)2 * BATCH * HWPX * 4;
  h16*   WSTEM  = (h16*)(ws + off);  off += (size_t)96 * KPAD * 2;
  h16*   WP1A   = (h16*)(ws + off);  off += (size_t)192 * 96 * 2;
  h16*   WP2A   = (h16*)(ws + off);  off += (size_t)96 * 192 * 2;
  h16*   WP1B   = (h16*)(ws + off);  off += (size_t)192 * 96 * 2;
  h16*   WP2B   = (h16*)(ws + off);  off += (size_t)96 * 192 * 2;
  h16*   WB2    = (h16*)(ws + off);  off += (size_t)9 * 96 * 96 * 2;
  h16*   WHEAD  = (h16*)(ws + off);  off += (size_t)9 * 16 * 96 * 2;

  float* OUT = (float*)d_out;

  // ---- pack weights, then TDM-warm them into L2
  pack_pw_kernel<<<(96 * KPAD + 255) / 256, 256, 0, stream>>>(stem_w, WSTEM, 96, 600, KPAD);
  pack_pw_kernel<<<(192 * 96 + 255) / 256, 256, 0, stream>>>(b1p1_w, WP1A, 192, 96, 96);
  pack_pw_kernel<<<(96 * 192 + 255) / 256, 256, 0, stream>>>(b1p2_w, WP2A, 96, 192, 192);
  pack_pw_kernel<<<(192 * 96 + 255) / 256, 256, 0, stream>>>(b3p1_w, WP1B, 192, 96, 96);
  pack_pw_kernel<<<(96 * 192 + 255) / 256, 256, 0, stream>>>(b3p2_w, WP2B, 96, 192, 192);
  pack_3x3_kernel<<<(9 * 96 * 96 + 255) / 256, 256, 0, stream>>>(b2_w, WB2, 96, 96, 96);
  pack_heads_kernel<<<(9 * 16 * 96 + 255) / 256, 256, 0, stream>>>(delta_w, ov_w, WHEAD);
  tdm_warm_kernel<<<1, 64, 0, stream>>>(WSTEM, TDMD);

  const int NPIX = BATCH * HWPX;
  const int NB   = (NPIX + 255) / 256;

  // ---- grid sample + input assembly + correlation
  sample_fill_kernel<<<NB, 256, 0, stream>>>(feat_A, feat_B, pwarp, pov, XIN, NORMS);
  {
    dim3 g(WW / 16, HH / 16, BATCH);
    size_t lds = (size_t)(16 * 16 * 128 + 24 * 24 * 128) * sizeof(h16);
    corr_kernel<<<g, 256, lds, stream>>>(XIN, NORMS);
  }

  const dim3 GG((HH / 8) * (WW / 16), 1, BATCH);   // 72 x 1 x 4
  auto launch_gn = [&](float* raw, int Cch, const float* gw, const float* gb,
                       const h16* res, h16* outh) {
    gn_stats_kernel<<<BATCH * 8, 256, 0, stream>>>(raw, Cch, STATS);
    size_t n = (size_t)BATCH * Cch * HWPX;
    gn_apply_kernel<<<(unsigned)((n + 255) / 256), 256, 0, stream>>>(
        raw, Cch, STATS, gw, gb, res, outh);
  };

  // ---- stem: 1x1, K=608 -> 96
  conv_gemm_wmma<6, 1><<<GG, 256, 0, stream>>>(XIN, KPAD, WSTEM, RAW, 1);
  launch_gn(RAW, 96, stem_gw, stem_gb, nullptr, ACT_A);

  // ---- block 1
  conv_gemm_wmma<12, 1><<<GG, 256, 0, stream>>>(ACT_A, 96, WP1A, RAW, 1);
  launch_gn(RAW, 192, b1p1_gw, b1p1_gb, nullptr, ACT192);
  dwconv3_kernel<<<(unsigned)(((size_t)BATCH * 192 * HWPX + 255) / 256), 256, 0, stream>>>(
      ACT192, b1dw_w, RAW, 192);
  launch_gn(RAW, 192, b1dw_gw, b1dw_gb, nullptr, ACT192);
  conv_gemm_wmma<6, 1><<<GG, 256, 0, stream>>>(ACT192, 192, WP2A, RAW, 1);
  launch_gn(RAW, 96, b1p2_gw, b1p2_gb, ACT_A, ACT_B);

  // ---- b2: 3x3 dilated-2, pad 2
  conv_gemm_wmma<6, 9><<<GG, 256, 0, stream>>>(ACT_B, 96, WB2, RAW, 2);
  launch_gn(RAW, 96, b2_gw, b2_gb, nullptr, ACT_C);

  // ---- block 3
  conv_gemm_wmma<12, 1><<<GG, 256, 0, stream>>>(ACT_C, 96, WP1B, RAW, 1);
  launch_gn(RAW, 192, b3p1_gw, b3p1_gb, nullptr, ACT192);
  dwconv3_kernel<<<(unsigned)(((size_t)BATCH * 192 * HWPX + 255) / 256), 256, 0, stream>>>(
      ACT192, b3dw_w, RAW, 192);
  launch_gn(RAW, 192, b3dw_gw, b3dw_gb, nullptr, ACT192);
  conv_gemm_wmma<6, 1><<<GG, 256, 0, stream>>>(ACT192, 192, WP2B, RAW, 1);
  launch_gn(RAW, 96, b3p2_gw, b3p2_gb, ACT_C, ACT_A);

  // ---- heads: 3x3 pad1, Cout padded to 16 (rows 0,1 = delta, row 2 = ov)
  conv_gemm_wmma<1, 9><<<GG, 256, 0, stream>>>(ACT_A, 96, WHEAD, RAW, 1);

  // ---- outputs
  final_kernel<<<NB, 256, 0, stream>>>(RAW, delta_b, ov_b, pwarp, pov, OUT);
}
